// GraphEncoder_41884521071102
// MI455X (gfx1250) — compile-verified, matrix operand order checked by reading, not run on hardware
//
#include <hip/hip_runtime.h>

#define NN   10000
#define EE   320000
#define GG   256
#define IND  78
#define HID  256
#define EMB  128
#define S    256          // fixed row stride for all node buffers
#define SCAN_T 1024

typedef __attribute__((ext_vector_type(16))) __bf16 v16bf;
typedef __attribute__((ext_vector_type(8)))  float  v8f;

__device__ __forceinline__ void atomAddF32(float* p, float v) {
  unsafeAtomicAdd(p, v);   // global_atomic_add_f32 (no-return)
}

// ---------- elementwise / conversion kernels ----------

// h[r*S + c] = c < din ? x[r*din + c] : 0
__global__ void k_pad_copy(const float* __restrict__ x, float* __restrict__ h,
                           int n, int din) {
  int i = blockIdx.x * blockDim.x + threadIdx.x;
  if (i >= n * S) return;
  int r = i >> 8, c = i & 255;
  h[i] = (c < din) ? x[r * din + c] : 0.f;
}

// Wt[o*S + k] = k < K ? W[k*O + o] : 0   (bf16, transposed + K-padded to 256)
__global__ void k_wt_bf16(const float* __restrict__ W, __bf16* __restrict__ Wt,
                          int K, int O) {
  int i = blockIdx.x * blockDim.x + threadIdx.x;
  if (i >= O * S) return;
  int o = i >> 8, k = i & 255;
  float v = (k < K) ? W[k * O + o] : 0.f;
  Wt[o * S + k] = (__bf16)v;
}

// bf16 convert with column zero-padding beyond din
__global__ void k_f2bf_pad(const float* __restrict__ src, __bf16* __restrict__ dst,
                           int n, int din) {
  int i = blockIdx.x * blockDim.x + threadIdx.x;
  if (i >= n) return;
  int c = i & 255;
  dst[i] = (c < din) ? (__bf16)src[i] : (__bf16)0.f;
}

__global__ void k_zero(float* __restrict__ p, int n) {
  int i = blockIdx.x * blockDim.x + threadIdx.x;
  if (i < n) p[i] = 0.f;
}

__global__ void k_zero_i(int* __restrict__ p, int n) {
  int i = blockIdx.x * blockDim.x + threadIdx.x;
  if (i < n) p[i] = 0;
}

// ---------- CSR-by-destination build (once per launch; edge_index is layer-invariant) ----

__global__ void k_count(const int* __restrict__ ei, int* __restrict__ deg, int E) {
  int e = blockIdx.x * blockDim.x + threadIdx.x;
  if (e < E) atomicAdd(&deg[ei[E + e]], 1);
}

// exclusive scan of deg[0..n) -> rowstart (and cursor copy); single 1024-thread block
__global__ void k_scan(const int* __restrict__ deg, int* __restrict__ rowstart,
                       int* __restrict__ cursor, int n) {
  __shared__ int sh[SCAN_T];
  int t = threadIdx.x;
  int per = (n + SCAN_T - 1) / SCAN_T;
  int lo = t * per, hi = min(lo + per, n);
  int s = 0;
  for (int v = lo; v < hi; ++v) s += deg[v];
  sh[t] = s;
  __syncthreads();
  for (int off = 1; off < SCAN_T; off <<= 1) {   // Hillis-Steele inclusive scan
    int val = (t >= off) ? sh[t - off] : 0;
    __syncthreads();
    if (t >= off) sh[t] += val;
    __syncthreads();
  }
  int base = (t == 0) ? 0 : sh[t - 1];
  for (int v = lo; v < hi; ++v) {
    rowstart[v] = base;
    cursor[v]   = base;
    base += deg[v];
  }
  if (t == SCAN_T - 1) rowstart[n] = base;
}

__global__ void k_scatter(const int* __restrict__ ei, int* __restrict__ cursor,
                          int* __restrict__ csr, int E) {
  int e = blockIdx.x * blockDim.x + threadIdx.x;
  if (e >= E) return;
  int pos = atomicAdd(&cursor[ei[E + e]], 1);
  csr[pos] = e;
}

// ---------- atomic-free fused aggregate ----------
// acc[v,c] = h[v,c] + sum_{e: dst(e)=v} relu(h[src(e),c] + ea[e,:]@We[:,c] + be[c])
// one block per node; one thread per feature column
__global__ void k_aggregate(const int* __restrict__ csr, const int* __restrict__ rowstart,
                            const int* __restrict__ ei, const float* __restrict__ ea,
                            const float* __restrict__ We, const float* __restrict__ be,
                            const float* __restrict__ h, float* __restrict__ acc, int din) {
  int v = blockIdx.x;
  int c = threadIdx.x;
  if (c >= din) return;
  float w0 = We[c], w1 = We[din + c], w2 = We[2 * din + c], w3 = We[3 * din + c];
  float bc = be[c];
  int p0 = rowstart[v], p1 = rowstart[v + 1];
  float sum = 0.f;
  for (int p = p0; p < p1; ++p) {
    int e   = csr[p];                        // block-uniform broadcast
    int src = ei[e];
    float4 a = *(const float4*)(ea + (size_t)e * 4);
    float elin = fmaf(a.x, w0, fmaf(a.y, w1, fmaf(a.z, w2, fmaf(a.w, w3, bc))));
    sum += fmaxf(h[src * S + c] + elin, 0.f);
  }
  acc[v * S + c] = h[v * S + c] + sum;
}

// ---------- WMMA bf16 GEMM: C[M,O] = act(A[M,256] @ Bt[O,256]^T + bias) ----------
template <int ACT>
__global__ void k_gemm_bf16(const __bf16* __restrict__ A, const __bf16* __restrict__ Bt,
                            const float* __restrict__ bias, float* __restrict__ C,
                            int M, int O, int ldc, int kchunks) {
  int gw   = (blockIdx.x * blockDim.x + threadIdx.x) >> 5;
  int lane = threadIdx.x & 31;
  int nt64 = O >> 6;
  int total = (M >> 4) * nt64;
  if (gw >= total) return;
  int mt = gw / nt64, nt = gw - mt * nt64;
  int m0 = mt << 4, n0 = nt << 6;
  int row = lane & 15, hi = lane >> 4;

  const __bf16* arow = A  + (m0 + row) * S + hi * 8;
  const __bf16* brow = Bt + (n0 + row) * S + hi * 8;

  v8f c0 = {}, c1 = {}, c2 = {}, c3 = {};
  for (int kc = 0; kc < kchunks; ++kc) {
    int kb = kc << 5;
    union Frag { v16bf v; uint4 q[2]; } af, b0, b1, b2, b3;
    af.q[0] = *(const uint4*)(arow + kb);
    af.q[1] = *(const uint4*)(arow + kb + 16);
    b0.q[0] = *(const uint4*)(brow + 0 * 16 * S + kb);
    b0.q[1] = *(const uint4*)(brow + 0 * 16 * S + kb + 16);
    b1.q[0] = *(const uint4*)(brow + 1 * 16 * S + kb);
    b1.q[1] = *(const uint4*)(brow + 1 * 16 * S + kb + 16);
    b2.q[0] = *(const uint4*)(brow + 2 * 16 * S + kb);
    b2.q[1] = *(const uint4*)(brow + 2 * 16 * S + kb + 16);
    b3.q[0] = *(const uint4*)(brow + 3 * 16 * S + kb);
    b3.q[1] = *(const uint4*)(brow + 3 * 16 * S + kb + 16);
    c0 = __builtin_amdgcn_wmma_f32_16x16x32_bf16(false, af.v, false, b0.v, (short)0, c0, false, false);
    c1 = __builtin_amdgcn_wmma_f32_16x16x32_bf16(false, af.v, false, b1.v, (short)0, c1, false, false);
    c2 = __builtin_amdgcn_wmma_f32_16x16x32_bf16(false, af.v, false, b2.v, (short)0, c2, false, false);
    c3 = __builtin_amdgcn_wmma_f32_16x16x32_bf16(false, af.v, false, b3.v, (short)0, c3, false, false);
  }
  int col = n0 + row;
#pragma unroll
  for (int i = 0; i < 8; ++i) {
    int r = m0 + i + hi * 8;
    float* crow = C + r * ldc;
    float v0 = c0[i] + bias[col];
    float v1 = c1[i] + bias[col + 16];
    float v2 = c2[i] + bias[col + 32];
    float v3 = c3[i] + bias[col + 48];
    if (ACT) { v0 = fmaxf(v0, 0.f); v1 = fmaxf(v1, 0.f); v2 = fmaxf(v2, 0.f); v3 = fmaxf(v3, 0.f); }
    crow[col] = v0; crow[col + 16] = v1; crow[col + 32] = v2; crow[col + 48] = v3;
  }
}

// ---------- BatchNorm (train-mode, biased var) ----------
__global__ void k_bnsum(const float* __restrict__ z, float* __restrict__ sums,
                        float* __restrict__ sqs, int nrows, int n) {
  int c = threadIdx.x;
  int r0 = blockIdx.x * nrows;
  int r1 = min(r0 + nrows, n);
  float s = 0.f, q = 0.f;
  for (int r = r0; r < r1; ++r) {
    float v = z[r * S + c];
    s += v; q += v * v;
  }
  atomAddF32(&sums[c], s);
  atomAddF32(&sqs[c], q);
}

__global__ void k_bnfin(const float* __restrict__ sums, const float* __restrict__ sqs,
                        const float* __restrict__ gamma, const float* __restrict__ beta,
                        float* __restrict__ scale, float* __restrict__ shift, int n) {
  int c = threadIdx.x;
  float inv = 1.f / (float)n;
  float mu  = sums[c] * inv;
  float var = sqs[c] * inv - mu * mu;
  float sc  = rsqrtf(var + 1e-5f) * gamma[c];
  scale[c] = sc;
  shift[c] = beta[c] - mu * sc;
}

__global__ void k_bnrelu(const float* __restrict__ z, const float* __restrict__ scale,
                         const float* __restrict__ shift, float* __restrict__ h, int n) {
  int i = blockIdx.x * blockDim.x + threadIdx.x;
  if (i >= n) return;
  int c = i & 255;
  h[i] = fmaxf(z[i] * scale[c] + shift[c], 0.f);
}

// ---------- global add pool ----------
__global__ void k_pool(const float* __restrict__ h, const int* __restrict__ batch,
                       float* __restrict__ g, int n) {
  int i = blockIdx.x * blockDim.x + threadIdx.x;
  if (i >= n * S) return;
  int r = i >> 8, c = i & 255;
  atomAddF32(&g[batch[r] * S + c], h[i]);
}

// ---------- launch ----------
static inline int cdiv(long a, long b) { return (int)((a + b - 1) / b); }

extern "C" void kernel_launch(void* const* d_in, const int* in_sizes, int n_in,
                              void* d_out, int out_size, void* d_ws, size_t ws_size,
                              hipStream_t stream) {
  (void)in_sizes; (void)n_in; (void)out_size; (void)ws_size;

  const float* x     = (const float*)d_in[0];
  const int*   ei    = (const int*)  d_in[1];
  const float* ea    = (const float*)d_in[2];
  const int*   batch = (const int*)  d_in[3];
  // params flattened in insertion order: per layer {We,be,W1,b1,W2,b2,gamma,beta}, then Wp,bp
  const float *We[3], *be_[3], *W1[3], *b1[3], *W2[3], *b2[3], *gamma[3], *beta[3];
  for (int l = 0; l < 3; ++l) {
    int base = 4 + l * 8;
    We[l]    = (const float*)d_in[base + 0];
    be_[l]   = (const float*)d_in[base + 1];
    W1[l]    = (const float*)d_in[base + 2];
    b1[l]    = (const float*)d_in[base + 3];
    W2[l]    = (const float*)d_in[base + 4];
    b2[l]    = (const float*)d_in[base + 5];
    gamma[l] = (const float*)d_in[base + 6];
    beta[l]  = (const float*)d_in[base + 7];
  }
  const float* Wp = (const float*)d_in[28];
  const float* bp = (const float*)d_in[29];

  // workspace layout
  float*  h     = (float*)d_ws;                    // [N,S]
  float*  acc   = h + (size_t)NN * S;              // [N,S]
  float*  t1    = acc + (size_t)NN * S;            // [N,S]
  float*  stats = t1 + (size_t)NN * S;             // sums|sqs|scale|shift (4x256)
  float*  sums  = stats, *sqs = stats + 256, *scale = stats + 512, *shift = stats + 768;
  float*  g     = stats + 1024;                    // [G,S]
  __bf16* abf   = (__bf16*)(g + (size_t)GG * S);   // [N,S]
  __bf16* wbf   = abf + (size_t)NN * S;            // 7 slots of 256*256
  __bf16* gbf   = wbf + (size_t)7 * 65536;         // [G,S]
  int*    deg   = (int*)(gbf + (size_t)GG * S);    // [N]
  int*    rowstart = deg + NN;                     // [N+1]
  int*    cursor   = rowstart + NN + 1;            // [N]
  int*    csr      = cursor + NN;                  // [E]

  // convert weights (transposed, K-padded to 256) once per launch
  for (int l = 0; l < 3; ++l) {
    k_wt_bf16<<<cdiv(256 * S, 256), 256, 0, stream>>>(W1[l], wbf + (size_t)(2 * l) * 65536,
                                                      l == 0 ? IND : HID, HID);
    k_wt_bf16<<<cdiv(256 * S, 256), 256, 0, stream>>>(W2[l], wbf + (size_t)(2 * l + 1) * 65536,
                                                      HID, HID);
  }
  k_wt_bf16<<<cdiv(EMB * S, 256), 256, 0, stream>>>(Wp, wbf + (size_t)6 * 65536, HID, EMB);

  // h = pad(x) to stride-256
  k_pad_copy<<<cdiv((long)NN * S, 256), 256, 0, stream>>>(x, h, NN, IND);

  // CSR by destination (edge_index is constant across layers)
  k_zero_i<<<cdiv(NN, 256), 256, 0, stream>>>(deg, NN);
  k_count<<<cdiv(EE, 256), 256, 0, stream>>>(ei, deg, EE);
  k_scan<<<1, SCAN_T, 0, stream>>>(deg, rowstart, cursor, NN);
  k_scatter<<<cdiv(EE, 256), 256, 0, stream>>>(ei, cursor, csr, EE);

  const int gemmBlocks = cdiv((long)(NN / 16) * (HID / 64) * 32, 256);  // 625*4 waves

  for (int l = 0; l < 3; ++l) {
    int din = (l == 0) ? IND : HID;
    int bdim = (din + 31) & ~31;
    // acc = h + scatter-free aggregate of relu(h[src] + edge_lin)
    k_aggregate<<<NN, bdim, 0, stream>>>(csr, rowstart, ei, ea, We[l], be_[l], h, acc, din);
    // t1 = relu(acc @ W1 + b1)   (WMMA bf16)
    k_f2bf_pad<<<cdiv((long)NN * S, 256), 256, 0, stream>>>(acc, abf, NN * S, din);
    k_gemm_bf16<1><<<gemmBlocks, 256, 0, stream>>>(abf, wbf + (size_t)(2 * l) * 65536,
                                                   b1[l], t1, NN, HID, HID, 8);
    // acc = t1 @ W2 + b2
    k_f2bf_pad<<<cdiv((long)NN * S, 256), 256, 0, stream>>>(t1, abf, NN * S, S);
    k_gemm_bf16<0><<<gemmBlocks, 256, 0, stream>>>(abf, wbf + (size_t)(2 * l + 1) * 65536,
                                                   b2[l], acc, NN, HID, HID, 8);
    // BN (train-mode) + relu -> h
    k_zero<<<2, 256, 0, stream>>>(sums, 512);
    k_bnsum<<<200, 256, 0, stream>>>(acc, sums, sqs, 50, NN);
    k_bnfin<<<1, 256, 0, stream>>>(sums, sqs, gamma[l], beta[l], scale, shift, NN);
    k_bnrelu<<<cdiv((long)NN * S, 256), 256, 0, stream>>>(acc, scale, shift, h, NN * S);
  }

  // global add pool -> g [G, 256]
  k_zero<<<cdiv((long)GG * S, 256), 256, 0, stream>>>(g, GG * S);
  k_pool<<<cdiv((long)NN * S, 256), 256, 0, stream>>>(h, batch, g, NN);

  // out = relu(g @ Wp + bp)  [G, EMB]
  k_f2bf_pad<<<cdiv((long)GG * S, 256), 256, 0, stream>>>(g, gbf, GG * S, S);
  k_gemm_bf16<1><<<cdiv((long)(GG / 16) * (EMB / 64) * 32, 256), 256, 0, stream>>>(
      gbf, wbf + (size_t)6 * 65536, bp, (float*)d_out, GG, EMB, EMB, 8);
}
